// VQVAE_43628277793247
// MI455X (gfx1250) — matrix operand order repaired and checked by reading, not compiled, not run on hardware
//
#include <hip/hip_runtime.h>
#include <hip/hip_bf16.h>
#include <stdint.h>
#include <stddef.h>

typedef __attribute__((ext_vector_type(16))) __bf16 v16bf;
typedef __attribute__((ext_vector_type(8)))  float  v8f;

__device__ __forceinline__ __bf16 to_bf(float f){ return (__bf16)f; }

// ---------------------------------------------------------------------------
// Implicit-GEMM conv, bf16 WMMA, f32 accumulate.
// X: NHWC f32 *interior* pointer of a halo-padded tensor. Halo guarantees all
// taps in-bounds -> no bounds checks; NHWC makes (dx,ci) contiguous so the A
// pointer walk is purely incremental (p += 32 per chunk, += rowSkip on tap-row
// wrap). Each wave computes NT adjacent 16-wide Cout tiles per A fetch to
// raise L2 arithmetic intensity (A amortized over NT WMMAs).
// ---------------------------------------------------------------------------
template<int NT>
__global__ void conv_gemm_wmma(
    const float* __restrict__ X, const __bf16* __restrict__ Wp,
    const float* __restrict__ bias, const float* __restrict__ Res,
    float* __restrict__ Y,
    int xRow, int xImg, int Cin,
    int WmSh, int HWmSh, int stride, int pad,
    int rowLen, int rowSkip,
    int Cout, int CoutPad, int yRow, int yImg,
    int sy, int oy, int sx, int ox, int K, int act, int M)
{
  const int lane = threadIdx.x & 31;
  const int half = lane >> 4, lr = lane & 15;
  const int mtile = blockIdx.x, ntile = blockIdx.y;
  int mrow = mtile * 16 + lr;
  int mr = mrow < M ? mrow : M - 1;
  int n  = mr >> HWmSh;
  int r2 = mr & ((1 << HWmSh) - 1);
  int ym = r2 >> WmSh;
  int xm = r2 & ((1 << WmSh) - 1);

  const float* p = X + (size_t)n * xImg;
  p += (ptrdiff_t)(ym * stride - pad) * xRow + (ptrdiff_t)(xm * stride - pad) * Cin + half * 8;
  const __bf16* bp = Wp + ntile * (NT * 16) + lr;

  v8f acc[NT];
#pragma unroll
  for (int t = 0; t < NT; ++t)
#pragma unroll
    for (int r = 0; r < 8; ++r) acc[t][r] = 0.f;

  int rc = 0;
  for (int kb = 0; kb < K; kb += 32) {
    v8f a0 = ((const v8f*)p)[0];
    v8f a1 = ((const v8f*)p)[2];
    __builtin_prefetch((const void*)(bp + (size_t)64 * CoutPad), 0, 0);
    v16bf a;
#pragma unroll
    for (int j = 0; j < 8; ++j) { a[j] = to_bf(a0[j]); a[8 + j] = to_bf(a1[j]); }
#pragma unroll
    for (int t = 0; t < NT; ++t) {
      v16bf b;
#pragma unroll
      for (int j = 0; j < 16; ++j) b[j] = bp[(size_t)(half * 8 + (j < 8 ? j : j + 8)) * CoutPad + t * 16];
      acc[t] = __builtin_amdgcn_wmma_f32_16x16x32_bf16(false, a, false, b,
                                                       (short)0, acc[t], false, false);
    }
    p += 32;
    bp += (size_t)32 * CoutPad;
    rc += 32;
    if (rc == rowLen) { rc = 0; p += rowSkip; }
  }
#pragma unroll
  for (int t = 0; t < NT; ++t) {
    int col = (ntile * NT + t) * 16 + lr;
    if (col < Cout) {
      float bv = bias ? bias[col] : 0.f;
#pragma unroll
      for (int r = 0; r < 8; ++r) {
        int row = mtile * 16 + r + 8 * half;
        if (row < M) {
          int n2 = row >> HWmSh;
          int q2 = row & ((1 << HWmSh) - 1);
          int y2 = q2 >> WmSh, x2 = q2 & ((1 << WmSh) - 1);
          size_t oidx = (size_t)n2 * yImg + (size_t)(y2 * sy + oy) * yRow
                      + (size_t)(x2 * sx + ox) * Cout + col;
          float v = acc[t][r] + bv;
          if (act == 1) v = v / (1.f + __expf(-v));
          if (Res) v += Res[oidx];
          Y[oidx] = v;
        }
      }
    }
  }
}

// ---------------------------------------------------------------------------
// Flash attention (unscaled softmax(q^T k), matching reference).
// Q,V: [Nb,S,c] f32; Kt: [Nb,c,S] f32; Z: [Nb,S,c]. c = CK*32 (32 or 64),
// templated so all fragment arrays have compile-time indices.
// ---------------------------------------------------------------------------
template<int CK>
__global__ void attn_flash(const float* __restrict__ Q, const float* __restrict__ Kt,
                           const float* __restrict__ V, float* __restrict__ Z,
                           int S, int logS)
{
  const int c = CK * 32;
  const int CN = CK * 2;
  const int lane = threadIdx.x & 31;
  const int half = lane >> 4, lr = lane & 15;
  int n  = blockIdx.x >> (logS - 4);
  int st = blockIdx.x & ((S >> 4) - 1);
  int s0 = st * 16;

  v16bf qf[CK];
#pragma unroll
  for (int kc = 0; kc < CK; ++kc) {
#pragma unroll
    for (int jj = 0; jj < 16; ++jj) {
      int kk = kc * 32 + half * 8 + (jj < 8 ? jj : jj + 8);
      qf[kc][jj] = to_bf(Q[((size_t)n * S + s0 + lr) * c + kk]);
    }
  }
  float mrow[8], lrow[8];
  v8f o[CN];
#pragma unroll
  for (int r = 0; r < 8; ++r) { mrow[r] = -3.0e38f; lrow[r] = 0.f; }
#pragma unroll
  for (int cn = 0; cn < CN; ++cn)
#pragma unroll
    for (int r = 0; r < 8; ++r) o[cn][r] = 0.f;

  __shared__ __bf16 Pl[16 * 32];

  for (int t0 = 0; t0 < S; t0 += 32) {
    v8f e0 = {0.f,0.f,0.f,0.f,0.f,0.f,0.f,0.f};
    v8f e1 = e0;
#pragma unroll
    for (int kc = 0; kc < CK; ++kc) {
      const float* kp = Kt + ((size_t)n * c + kc * 32) * S + t0 + lr;
      v16bf b0, b1;
#pragma unroll
      for (int j = 0; j < 16; ++j) {
        int kk = half * 8 + (j < 8 ? j : j + 8);
        b0[j] = to_bf(kp[(size_t)kk * S]); b1[j] = to_bf(kp[(size_t)kk * S + 16]);
      }
      e0 = __builtin_amdgcn_wmma_f32_16x16x32_bf16(false, qf[kc], false, b0, (short)0, e0, false, false);
      e1 = __builtin_amdgcn_wmma_f32_16x16x32_bf16(false, qf[kc], false, b1, (short)0, e1, false, false);
    }
#pragma unroll
    for (int r = 0; r < 8; ++r) {
      float tm = fmaxf(e0[r], e1[r]);
      for (int m = 1; m < 16; m <<= 1) tm = fmaxf(tm, __shfl_xor(tm, m, 32));
      float nm = fmaxf(mrow[r], tm);
      float scale = __expf(mrow[r] - nm);
      float p0 = __expf(e0[r] - nm);
      float p1 = __expf(e1[r] - nm);
      float ps = p0 + p1;
      for (int m = 1; m < 16; m <<= 1) ps += __shfl_xor(ps, m, 32);
      lrow[r] = lrow[r] * scale + ps;
      mrow[r] = nm;
#pragma unroll
      for (int cn = 0; cn < CN; ++cn) o[cn][r] *= scale;
      int row = r + 8 * half;
      Pl[row * 32 + lr]      = to_bf(p0);
      Pl[row * 32 + 16 + lr] = to_bf(p1);
    }
    asm volatile("s_wait_dscnt 0" ::: "memory");
    v16bf pf;
#pragma unroll
    for (int jj = 0; jj < 16; ++jj) {
      int tl = half * 8 + (jj < 8 ? jj : jj + 8);
      pf[jj] = Pl[lr * 32 + tl];
    }
#pragma unroll
    for (int cn = 0; cn < CN; ++cn) {
      const float* vp = V + ((size_t)n * S + t0) * c + cn * 16 + lr;
      v16bf bv;
#pragma unroll
      for (int j = 0; j < 16; ++j) { int tk = half * 8 + (j < 8 ? j : j + 8); bv[j] = to_bf(vp[(size_t)tk * c]); }
      o[cn] = __builtin_amdgcn_wmma_f32_16x16x32_bf16(false, pf, false, bv, (short)0, o[cn], false, false);
    }
    asm volatile("s_wait_dscnt 0" ::: "memory");
  }
#pragma unroll
  for (int r = 0; r < 8; ++r) {
    int row = s0 + r + 8 * half;
    float inv = 1.f / lrow[r];
#pragma unroll
    for (int cn = 0; cn < CN; ++cn)
      Z[((size_t)n * S + row) * c + cn * 16 + lr] = o[cn][r] * inv;
  }
}

// ------------------------------- helpers -----------------------------------

__global__ void pack_conv_w_k(const float* __restrict__ w, __bf16* __restrict__ out,
                              int CinPad, int CinReal, int Cout, int kh, int kw,
                              int Kpad, int CoutPad)
{
  int idx = blockIdx.x * 256 + threadIdx.x;
  int total = Kpad * CoutPad;
  if (idx >= total) return;
  int k = idx / CoutPad, co = idx - k * CoutPad;
  int ci = k % CinPad; int e = k / CinPad;
  int dy = e / kw, dx = e - (e / kw) * kw;
  float v = 0.f;
  if (ci < CinReal && co < Cout && e < kh * kw)
    v = w[(((size_t)co * CinReal + ci) * kh + dy) * kw + dx];
  out[idx] = to_bf(v);
}

__global__ void pack_convT_w_k(const float* __restrict__ w, __bf16* __restrict__ out,
                               int Cin, int Cout, int CoutPad)
{
  int idx = blockIdx.x * 256 + threadIdx.x;
  int total = 4 * Cin * CoutPad;
  if (idx >= total) return;
  int tu = idx / (Cin * CoutPad);
  int rem = idx - tu * (Cin * CoutPad);
  int ci = rem / CoutPad, co = rem - (rem / CoutPad) * CoutPad;
  int t = tu >> 1, u = tu & 1;
  float v = 0.f;
  if (co < Cout)
    v = w[(((size_t)co * Cin + ci) * 2 + (1 - t)) * 2 + (1 - u)];
  out[idx] = to_bf(v);
}

__global__ void pack_cbT_k(const float* __restrict__ cb, __bf16* __restrict__ out)
{
  int idx = blockIdx.x * 256 + threadIdx.x;
  if (idx >= 128 * 128) return;
  int k = idx >> 7, j = idx & 127;
  out[idx] = to_bf(cb[(size_t)j * 128 + k]);
}

__global__ void cbnorm_k(const float* __restrict__ cb, float* __restrict__ cbn)
{
  int j = threadIdx.x;
  if (j >= 128) return;
  float s = 0.f;
  for (int k = 0; k < 128; ++k) { float v = cb[(size_t)j * 128 + k]; s += v * v; }
  cbn[j] = s;
}

// zero the 3-pixel halo of a padded [N, H+6, W+6, C] tensor (interior kept)
__global__ void zero_halo_k(float* __restrict__ buf, int H, int W, int logC, int total)
{
  int idx = blockIdx.x * 256 + threadIdx.x;
  if (idx >= total) return;
  int sp = idx >> logC;
  int Wp = W + 6;
  int xx = sp % Wp;
  int t = sp / Wp;
  int yy = t % (H + 6);
  if (xx < 3 || xx >= W + 3 || yy < 3 || yy >= H + 3) buf[idx] = 0.f;
}

__global__ void gn_stats_k(const float* __restrict__ X, float* __restrict__ mv,
                           int HW, int C, int logcs, int logW, int row, int img)
{
  int ng = blockIdx.x; int n = ng >> 5; int g = ng & 31;
  int cs = 1 << logcs; int c0 = g << logcs;
  int total = HW << logcs;
  float s = 0.f, ss = 0.f;
  for (int t = threadIdx.x; t < total; t += 256) {
    int sp = t >> logcs; int i = t & (cs - 1);
    int y = sp >> logW; int x = sp & ((1 << logW) - 1);
    float v = X[(size_t)n * img + (size_t)y * row + x * C + c0 + i];
    s += v; ss += v * v;
  }
  __shared__ float sh1[256], sh2[256];
  sh1[threadIdx.x] = s; sh2[threadIdx.x] = ss;
  __syncthreads();
  for (int st = 128; st > 0; st >>= 1) {
    if (threadIdx.x < st) { sh1[threadIdx.x] += sh1[threadIdx.x + st]; sh2[threadIdx.x] += sh2[threadIdx.x + st]; }
    __syncthreads();
  }
  if (threadIdx.x == 0) {
    float m = sh1[0] / total;
    mv[ng * 2] = m; mv[ng * 2 + 1] = sh2[0] / total - m * m;
  }
}

__global__ void gn_apply_k(const float* __restrict__ X, const float* __restrict__ mv,
                           const float* __restrict__ gamma, const float* __restrict__ beta,
                           float* __restrict__ Y, int logC, int logW, int logH,
                           int row, int img, int total)
{
  int idx = blockIdx.x * 256 + threadIdx.x;
  if (idx >= total) return;
  int C = 1 << logC;
  int c = idx & (C - 1);
  int sp = idx >> logC;
  int x = sp & ((1 << logW) - 1);
  int y = (sp >> logW) & ((1 << logH) - 1);
  int n = sp >> (logW + logH);
  size_t off = (size_t)n * img + (size_t)y * row + x * C + c;
  int g = c >> (logC - 5);
  float m = mv[(n * 32 + g) * 2], v = mv[(n * 32 + g) * 2 + 1];
  Y[off] = (X[off] - m) * rsqrtf(v + 1e-5f) * gamma[c] + beta[c];
}

__global__ void silu_k(const float* __restrict__ X, float* __restrict__ Y, int total)
{
  int idx = blockIdx.x * 256 + threadIdx.x;
  if (idx >= total) return;
  float v = X[idx];
  Y[idx] = v / (1.f + __expf(-v));
}

// padded interior -> flat, with SiLU
__global__ void silu_from_pad_k(const float* __restrict__ X, float* __restrict__ Y,
                                int logC, int logW, int logH, int row, int img, int total)
{
  int idx = blockIdx.x * 256 + threadIdx.x;
  if (idx >= total) return;
  int C = 1 << logC;
  int c = idx & (C - 1);
  int sp = idx >> logC;
  int x = sp & ((1 << logW) - 1);
  int y = (sp >> logW) & ((1 << logH) - 1);
  int n = sp >> (logW + logH);
  float v = X[(size_t)n * img + (size_t)y * row + x * C + c];
  Y[idx] = v / (1.f + __expf(-v));
}

__global__ void transpose_sc_k(const float* __restrict__ in, float* __restrict__ out,
                               int logS, int logc, int total)
{
  int idx = blockIdx.x * 256 + threadIdx.x;
  if (idx >= total) return;
  int ch = idx & ((1 << logc) - 1);
  int t = idx >> logc;
  int s = t & ((1 << logS) - 1);
  int n = t >> logS;
  out[((((size_t)n << logc) + ch) << logS) + s] = in[idx];
}

// NCHW [4,3,128,128] -> padded-interior NHWC [4,128,128,32] (pad channels 0)
__global__ void nchw2nhwc_pad32_k(const float* __restrict__ in, float* __restrict__ out,
                                  int row, int img)
{
  int idx = blockIdx.x * 256 + threadIdx.x;
  if (idx >= 4 * 128 * 128 * 32) return;
  int c = idx & 31; int sp = idx >> 5;
  int x = sp & 127; int y = (sp >> 7) & 127; int n = sp >> 14;
  out[(size_t)n * img + (size_t)y * row + x * 32 + c] =
      (c < 3) ? in[(((size_t)n * 3 + c) * 128 + y) * 128 + x] : 0.f;
}

// flat NHWC [4,128,128,3] -> NCHW [4,3,128,128]
__global__ void nhwc2nchw_k(const float* __restrict__ in, float* __restrict__ out,
                            int N, int C, int H, int W)
{
  int idx = blockIdx.x * 256 + threadIdx.x;
  int total = N * C * H * W;
  if (idx >= total) return;
  int w = idx % W; int t = idx / W; int h = t % H; t /= H; int c = t % C; int n = t / C;
  out[idx] = in[(((size_t)n * H + h) * W + w) * C + c];
}

__global__ void vq_argmin_gather_k(const float* __restrict__ G, const float* __restrict__ cbn,
                                   const float* __restrict__ cb, float* __restrict__ qz,
                                   int row, int img)
{
  int m = blockIdx.x * 64 + threadIdx.x;
  if (m >= 4096) return;
  float best = 3.0e38f; int bid = 0;
  for (int j = 0; j < 128; ++j) {
    float d = cbn[j] - 2.f * G[(size_t)m * 128 + j];
    if (d < best) { best = d; bid = j; }
  }
  int x = m & 31, y = (m >> 5) & 31, n = m >> 10;
  float* q = qz + (size_t)n * img + (size_t)y * row + x * 128;
  for (int k = 0; k < 128; ++k) q[k] = cb[(size_t)bid * 128 + k];
}

__global__ void zero1_k(float* p) { p[0] = 0.f; }

__global__ void vq_loss_k(const float* __restrict__ qz, const float* __restrict__ z,
                          float* __restrict__ out, int row, int img, float scale)
{
  __shared__ float sh[256];
  float v = 0.f;
  for (int idx = blockIdx.x * 256 + threadIdx.x; idx < 524288; idx += gridDim.x * 256) {
    int c = idx & 127; int sp = idx >> 7;
    int x = sp & 31; int y = (sp >> 5) & 31; int n = sp >> 10;
    size_t off = (size_t)n * img + (size_t)y * row + x * 128 + c;
    float d = qz[off] - z[off]; v += d * d;
  }
  sh[threadIdx.x] = v; __syncthreads();
  for (int s = 128; s > 0; s >>= 1) {
    if (threadIdx.x < s) sh[threadIdx.x] += sh[threadIdx.x + s];
    __syncthreads();
  }
  if (threadIdx.x == 0) atomicAdd(out, sh[0] * scale);
}

// ------------------------------ host side ----------------------------------

struct ConvP { const float* w; const float* b; };
struct GnP   { const float* g; const float* b; };
struct ResP  { GnP gn; ConvP c1, c2; };
struct AttnP { GnP gn; ConvP q, k, v, o; };
struct EncP  { ConvP cin, down1, down2; GnP gn1, gn2; ResP r[10]; AttnP a1, a2; };
struct DecP  { ConvP up1, up2, outc; GnP gn1, gn2; ResP r[10]; AttnP a1, a2; };

struct Walk {
  void* const* din; int idx; bool sorted;
  const float* nx() { return (const float*)din[idx++]; }
};
static void rdConv(Walk& w, ConvP& c){ if (w.sorted){ c.b=w.nx(); c.w=w.nx(); } else { c.w=w.nx(); c.b=w.nx(); } }
static void rdGn  (Walk& w, GnP& g){ if (w.sorted){ g.b=w.nx(); g.g=w.nx(); } else { g.g=w.nx(); g.b=w.nx(); } }
static void rdRes (Walk& w, ResP& r){
  if (w.sorted){ rdConv(w,r.c1); rdConv(w,r.c2); rdGn(w,r.gn); }
  else         { rdGn(w,r.gn); rdConv(w,r.c1); rdConv(w,r.c2); }
}
static void rdAttn(Walk& w, AttnP& a){
  if (w.sorted){ rdGn(w,a.gn); rdConv(w,a.k); rdConv(w,a.o); rdConv(w,a.q); rdConv(w,a.v); }
  else         { rdGn(w,a.gn); rdConv(w,a.q); rdConv(w,a.k); rdConv(w,a.v); rdConv(w,a.o); }
}
static void rdEnc(Walk& w, EncP& e){
  if (w.sorted){
    rdAttn(w,e.a1); rdAttn(w,e.a2); rdConv(w,e.down1); rdConv(w,e.down2);
    rdGn(w,e.gn1); rdGn(w,e.gn2); rdConv(w,e.cin);
    for (int i=1;i<=9;++i) rdRes(w,e.r[i]);
  } else {
    rdConv(w,e.cin); rdRes(w,e.r[1]); rdRes(w,e.r[2]); rdRes(w,e.r[3]);
    rdGn(w,e.gn1); rdConv(w,e.down1); rdRes(w,e.r[4]); rdRes(w,e.r[5]);
    rdAttn(w,e.a1); rdRes(w,e.r[6]); rdGn(w,e.gn2); rdConv(w,e.down2);
    rdRes(w,e.r[7]); rdRes(w,e.r[8]); rdAttn(w,e.a2); rdRes(w,e.r[9]);
  }
}
static void rdDec(Walk& w, DecP& d){
  if (w.sorted){
    rdAttn(w,d.a1); rdAttn(w,d.a2); rdGn(w,d.gn1); rdGn(w,d.gn2); rdConv(w,d.outc);
    for (int i=1;i<=9;++i) rdRes(w,d.r[i]);
    rdConv(w,d.up1); rdConv(w,d.up2);
  } else {
    rdRes(w,d.r[1]); rdRes(w,d.r[2]); rdAttn(w,d.a1); rdRes(w,d.r[3]);
    rdGn(w,d.gn1); rdConv(w,d.up1); rdRes(w,d.r[4]); rdRes(w,d.r[5]);
    rdAttn(w,d.a2); rdRes(w,d.r[6]); rdGn(w,d.gn2); rdConv(w,d.up2);
    rdRes(w,d.r[7]); rdRes(w,d.r[8]); rdRes(w,d.r[9]); rdConv(w,d.outc);
  }
}

struct PackedW { __bf16* wp; int Kpad; int CoutPad; };
struct ResPk { PackedW c1, c2; };
struct AttnPk { PackedW q, k, v, o; };

struct TD { float* base; float* in; int row, img, C, H, W; };

static inline int cdiv(int a, int b){ return (a + b - 1) / b; }

extern "C" void kernel_launch(void* const* d_in, const int* in_sizes, int n_in,
                              void* d_out, int out_size, void* d_ws, size_t ws_size,
                              hipStream_t stream)
{
  (void)n_in; (void)out_size; (void)ws_size;
  const int N = 4;
  bool sorted = (in_sizes[0] != 196608);
  Walk w{d_in, 0, sorted};
  EncP enc; DecP dec; const float* cb = nullptr; const float* x = nullptr;
  if (sorted) { cb = w.nx(); rdDec(w, dec); rdEnc(w, enc); x = w.nx(); }
  else        { x = w.nx(); rdEnc(w, enc); rdDec(w, dec); cb = w.nx(); }

  // ---- workspace ----
  char* base = (char*)d_ws; size_t bo = 0;
  auto alloc = [&](size_t bytes) -> void* {
    bo = (bo + 63) & ~(size_t)63; void* p = base + bo; bo += bytes; return p;
  };
  const size_t PADSZ = (size_t)4 * 134 * 134 * 32;
  float* xh   = (float*)alloc(PADSZ * 4);
  float* BA   = (float*)alloc(PADSZ * 4);
  float* BB   = (float*)alloc(PADSZ * 4);
  float* BC   = (float*)alloc(PADSZ * 4);
  float* QZ   = (float*)alloc(PADSZ * 4);
  float* FS   = (float*)alloc((size_t)2097152 * 4);
  float* attq = (float*)alloc(524288 * 4);
  float* attk = (float*)alloc(524288 * 4);
  float* attv = (float*)alloc(524288 * 4);
  float* attkt= (float*)alloc(524288 * 4);
  float* attz = (float*)alloc(524288 * 4);
  float* attzs= (float*)alloc(524288 * 4);
  float* Gbuf = (float*)alloc(524288 * 4);
  float* cbnB = (float*)alloc(128 * 4);
  float* mv   = (float*)alloc(256 * 4);
  float* outf = (float*)d_out;

  auto mkpad = [&](float* buf, int H, int W, int C) -> TD {
    int row = (W + 6) * C;
    return TD{ buf, buf + 3 * row + 3 * C, row, (H + 6) * row, C, H, W };
  };
  auto mkflat = [&](float* buf, int H, int W, int C) -> TD {
    int row = W * C;
    return TD{ buf, buf, row, H * row, C, H, W };
  };

  // ---- weight packing ----
  auto packConv = [&](const ConvP& cp, int cinPad, int cinReal, int cout, int kk) -> PackedW {
    int Kpad = kk * kk * cinPad;
    int CoutPad = cdiv(cout, 16) * 16;
    __bf16* wp = (__bf16*)alloc((size_t)Kpad * CoutPad * 2);
    int total = Kpad * CoutPad;
    pack_conv_w_k<<<cdiv(total, 256), 256, 0, stream>>>(cp.w, wp, cinPad, cinReal, cout, kk, kk, Kpad, CoutPad);
    return {wp, Kpad, CoutPad};
  };
  auto packConvT = [&](const ConvP& cp, int cin, int cout) -> PackedW {
    int CoutPad = cdiv(cout, 16) * 16;
    __bf16* wp = (__bf16*)alloc((size_t)4 * cin * CoutPad * 2);
    int total = 4 * cin * CoutPad;
    pack_convT_w_k<<<cdiv(total, 256), 256, 0, stream>>>(cp.w, wp, cin, cout, CoutPad);
    return {wp, cin, CoutPad};
  };
  auto packRes = [&](const ResP& rp, int C) -> ResPk {
    return { packConv(rp.c1, C, C, C, 7), packConv(rp.c2, C, C, C, 1) };
  };
  auto packAttn = [&](const AttnP& ap, int C) -> AttnPk {
    int hc = C / 2;
    return { packConv(ap.q, C, C, hc, 1), packConv(ap.k, C, C, hc, 1),
             packConv(ap.v, C, C, hc, 1), packConv(ap.o, hc, hc, C, 1) };
  };

  const int encC[10] = {0,32,32,32,64,64,64,128,128,128};
  const int decC[10] = {0,128,128,128,64,64,64,32,32,32};
  PackedW p_in = packConv(enc.cin, 32, 3, 32, 7);
  PackedW p_d1 = packConv(enc.down1, 32, 32, 64, 2);
  PackedW p_d2 = packConv(enc.down2, 64, 64, 128, 2);
  ResPk  ePk[10]; for (int i = 1; i <= 9; ++i) ePk[i] = packRes(enc.r[i], encC[i]);
  AttnPk eA1 = packAttn(enc.a1, 64), eA2 = packAttn(enc.a2, 128);
  ResPk  dPk[10]; for (int i = 1; i <= 9; ++i) dPk[i] = packRes(dec.r[i], decC[i]);
  AttnPk dA1 = packAttn(dec.a1, 128), dA2 = packAttn(dec.a2, 64);
  PackedW p_u1 = packConvT(dec.up1, 128, 64);
  PackedW p_u2 = packConvT(dec.up2, 64, 32);
  PackedW p_out = packConv(dec.outc, 32, 32, 3, 1);
  PackedW p_cb;
  {
    __bf16* wp = (__bf16*)alloc(128 * 128 * 2);
    pack_cbT_k<<<cdiv(16384, 256), 256, 0, stream>>>(cb, wp);
    p_cb = {wp, 128, 128};
  }
  cbnorm_k<<<1, 128, 0, stream>>>(cb, cbnB);

  // ---- op helpers ----
  auto zhalo = [&](const TD& T) {
    int total = N * (T.H + 6) * (T.W + 6) * T.C;
    zero_halo_k<<<cdiv(total, 256), 256, 0, stream>>>(T.base, T.H, T.W, __builtin_ctz(T.C), total);
  };
  auto convN = [&](const TD& Xd, const PackedW& pw, const float* bias, const float* ResIn,
                   const TD& Yd, int kh, int stride, int pad, int Cout, int act,
                   int sy, int oy, int sx, int ox, const __bf16* wp, int K) {
    int Hm = (Xd.H + 2 * pad - kh) / stride + 1;
    int Wm = (Xd.W + 2 * pad - kh) / stride + 1;
    int M = N * Hm * Wm;
    int NT = (pw.CoutPad % 64 == 0) ? 4 : (pw.CoutPad % 32 == 0 ? 2 : 1);
    dim3 g(cdiv(M, 16), pw.CoutPad / (16 * NT));
    int wsh = __builtin_ctz(Wm), hwsh = __builtin_ctz(Hm * Wm);
    if (NT == 4)
      conv_gemm_wmma<4><<<g, 32, 0, stream>>>(Xd.in, wp, bias, ResIn, Yd.in,
          Xd.row, Xd.img, Xd.C, wsh, hwsh, stride, pad, kh * Xd.C, Xd.row - kh * Xd.C,
          Cout, pw.CoutPad, Yd.row, Yd.img, sy, oy, sx, ox, K, act, M);
    else if (NT == 2)
      conv_gemm_wmma<2><<<g, 32, 0, stream>>>(Xd.in, wp, bias, ResIn, Yd.in,
          Xd.row, Xd.img, Xd.C, wsh, hwsh, stride, pad, kh * Xd.C, Xd.row - kh * Xd.C,
          Cout, pw.CoutPad, Yd.row, Yd.img, sy, oy, sx, ox, K, act, M);
    else
      conv_gemm_wmma<1><<<g, 32, 0, stream>>>(Xd.in, wp, bias, ResIn, Yd.in,
          Xd.row, Xd.img, Xd.C, wsh, hwsh, stride, pad, kh * Xd.C, Xd.row - kh * Xd.C,
          Cout, pw.CoutPad, Yd.row, Yd.img, sy, oy, sx, ox, K, act, M);
  };
  auto conv = [&](const TD& Xd, const PackedW& pw, const float* bias, const float* ResIn,
                  const TD& Yd, int kh, int stride, int pad, int Cout, int act) {
    convN(Xd, pw, bias, ResIn, Yd, kh, stride, pad, Cout, act, 1, 0, 1, 0, pw.wp, pw.Kpad);
  };
  auto convT = [&](const TD& Xd, const PackedW& pw, const float* bias, const TD& Yd, int Cout) {
    for (int t = 0; t < 2; ++t)
      for (int u = 0; u < 2; ++u)
        convN(Xd, pw, bias, nullptr, Yd, 1, 1, 0, Cout, 0, 2, t, 2, u,
              pw.wp + (size_t)(t * 2 + u) * Xd.C * pw.CoutPad, Xd.C);
  };
  auto gnorm = [&](const TD& Xd, const GnP& g, const TD& Yd) {
    int C = Xd.C;
    gn_stats_k<<<N * 32, 256, 0, stream>>>(Xd.in, mv, Xd.H * Xd.W, C,
        __builtin_ctz(C) - 5, __builtin_ctz(Xd.W), Xd.row, Xd.img);
    int total = N * Xd.H * Xd.W * C;
    gn_apply_k<<<cdiv(total, 256), 256, 0, stream>>>(Xd.in, mv, g.g, g.b, Yd.in,
        __builtin_ctz(C), __builtin_ctz(Xd.W), __builtin_ctz(Xd.H), Xd.row, Xd.img, total);
  };
  auto resblock = [&](const TD& cur, const ResP& rp, const ResPk& pk, const TD& t1, const TD& t2) {
    gnorm(cur, rp.gn, t1);
    zhalo(t1);
    conv(t1, pk.c1, rp.c1.b, nullptr, t2, 7, 1, 3, cur.C, 1);
    conv(t2, pk.c2, rp.c2.b, cur.in, cur, 1, 1, 0, cur.C, 0);
  };
  auto attention = [&](const TD& cur, const AttnP& ap, const AttnPk& pk, const TD& t1) {
    int hc = cur.C / 2, S = cur.H * cur.W;
    TD qd = mkflat(attq, cur.H, cur.W, hc), kd = mkflat(attk, cur.H, cur.W, hc);
    TD vd = mkflat(attv, cur.H, cur.W, hc), zd = mkflat(attzs, cur.H, cur.W, hc);
    gnorm(cur, ap.gn, t1);
    conv(t1, pk.q, ap.q.b, nullptr, qd, 1, 1, 0, hc, 0);
    conv(t1, pk.k, ap.k.b, nullptr, kd, 1, 1, 0, hc, 0);
    conv(t1, pk.v, ap.v.b, nullptr, vd, 1, 1, 0, hc, 0);
    int tt = N * S * hc;
    transpose_sc_k<<<cdiv(tt, 256), 256, 0, stream>>>(attk, attkt,
        __builtin_ctz(S), __builtin_ctz(hc), tt);
    if (hc == 32)
      attn_flash<1><<<N * (S / 16), 32, 0, stream>>>(attq, attkt, attv, attz,
          S, __builtin_ctz(S));
    else
      attn_flash<2><<<N * (S / 16), 32, 0, stream>>>(attq, attkt, attv, attz,
          S, __builtin_ctz(S));
    silu_k<<<cdiv(tt, 256), 256, 0, stream>>>(attz, attzs, tt);
    conv(zd, pk.o, ap.o.b, cur.in, cur, 1, 1, 0, cur.C, 0);
  };

  // ---- tensor views ----
  TD xh1 = mkpad(xh, 128, 128, 32);
  TD BA1 = mkpad(BA, 128, 128, 32), BB1 = mkpad(BB, 128, 128, 32), BC1 = mkpad(BC, 128, 128, 32);
  TD BA2 = mkpad(BA, 64, 64, 64),   BB2 = mkpad(BB, 64, 64, 64),   BC2 = mkpad(BC, 64, 64, 64);
  TD BA3 = mkpad(BA, 32, 32, 128),  BB3 = mkpad(BB, 32, 32, 128),  BC3 = mkpad(BC, 32, 32, 128);
  TD QZ3 = mkpad(QZ, 32, 32, 128);
  TD Gf  = mkflat(Gbuf, 32, 32, 128);
  TD FSf = mkflat(FS, 128, 128, 32);
  TD OUTf = mkflat(attq, 128, 128, 3);

  // ---- forward ----
  nchw2nhwc_pad32_k<<<cdiv(2097152, 256), 256, 0, stream>>>(x, xh1.in, xh1.row, xh1.img);
  zhalo(xh1);

  // encoder
  conv(xh1, p_in, enc.cin.b, nullptr, BA1, 7, 1, 3, 32, 0);
  resblock(BA1, enc.r[1], ePk[1], BB1, BC1);
  resblock(BA1, enc.r[2], ePk[2], BB1, BC1);
  resblock(BA1, enc.r[3], ePk[3], BB1, BC1);
  gnorm(BA1, enc.gn1, BB1);
  conv(BB1, p_d1, enc.down1.b, nullptr, BC2, 2, 2, 0, 64, 0);
  resblock(BC2, enc.r[4], ePk[4], BA2, BB2);
  resblock(BC2, enc.r[5], ePk[5], BA2, BB2);
  attention(BC2, enc.a1, eA1, BA2);
  resblock(BC2, enc.r[6], ePk[6], BA2, BB2);
  gnorm(BC2, enc.gn2, BA2);
  conv(BA2, p_d2, enc.down2.b, nullptr, BB3, 2, 2, 0, 128, 0);
  resblock(BB3, enc.r[7], ePk[7], BA3, BC3);
  resblock(BB3, enc.r[8], ePk[8], BA3, BC3);
  attention(BB3, enc.a2, eA2, BA3);
  resblock(BB3, enc.r[9], ePk[9], BA3, BC3);

  // VQ
  conv(BB3, p_cb, nullptr, nullptr, Gf, 1, 1, 0, 128, 0);
  vq_argmin_gather_k<<<cdiv(4096, 64), 64, 0, stream>>>(Gbuf, cbnB, cb, QZ3.in, QZ3.row, QZ3.img);
  zero1_k<<<1, 1, 0, stream>>>(outf + 196608);
  vq_loss_k<<<256, 256, 0, stream>>>(QZ3.in, BB3.in, outf + 196608, QZ3.row, QZ3.img,
                                     1.25f / 524288.f);

  // decoder
  resblock(QZ3, dec.r[1], dPk[1], BA3, BC3);
  resblock(QZ3, dec.r[2], dPk[2], BA3, BC3);
  attention(QZ3, dec.a1, dA1, BA3);
  resblock(QZ3, dec.r[3], dPk[3], BA3, BC3);
  gnorm(QZ3, dec.gn1, BA3);
  convT(BA3, p_u1, dec.up1.b, BB2, 64);
  resblock(BB2, dec.r[4], dPk[4], BA2, BC2);
  resblock(BB2, dec.r[5], dPk[5], BA2, BC2);
  attention(BB2, dec.a2, dA2, BA2);
  resblock(BB2, dec.r[6], dPk[6], BA2, BC2);
  gnorm(BB2, dec.gn2, BA2);
  convT(BA2, p_u2, dec.up2.b, BC1, 32);
  resblock(BC1, dec.r[7], dPk[7], BA1, BB1);
  resblock(BC1, dec.r[8], dPk[8], BA1, BB1);
  resblock(BC1, dec.r[9], dPk[9], BA1, BB1);
  silu_from_pad_k<<<cdiv(2097152, 256), 256, 0, stream>>>(BC1.in, FS,
      5, 7, 7, BC1.row, BC1.img, 2097152);
  conv(FSf, p_out, dec.outc.b, nullptr, OUTf, 1, 1, 0, 3, 0);
  nhwc2nchw_k<<<cdiv(196608, 256), 256, 0, stream>>>(attq, outf, N, 3, 128, 128);
}
